// VectorQuantizer_25615184953910
// MI455X (gfx1250) — compile-verified, matrix operand order checked by reading, not run on hardware
//
#include <hip/hip_runtime.h>
#include <hip/hip_bf16.h>

#define N_E 512
#define E_DIM 256
#define NEXP 11          // 10 experts + shared
#define BTOK 8192
#define SK_ITERS 100
#define SKB 16           // blocks per expert in sinkhorn persistent kernel

typedef __bf16 bf16_t;
typedef __attribute__((ext_vector_type(16))) __bf16 v16bf;
typedef __attribute__((ext_vector_type(8)))  float  v8f;

// ---------- helpers ----------
__device__ inline unsigned f2u(float f){
  unsigned u = __float_as_uint(f);
  return (u & 0x80000000u) ? ~u : (u | 0x80000000u);
}
__device__ inline float u2f(unsigned u){
  u = (u & 0x80000000u) ? (u & 0x7fffffffu) : ~u;
  return __uint_as_float(u);
}

// per-expert software grid barrier (agent scope)
__device__ inline void gbar(unsigned* cnt, unsigned* gen, unsigned nblk){
  __threadfence();
  __syncthreads();
  if (threadIdx.x == 0){
    unsigned g0 = __hip_atomic_load(gen, __ATOMIC_ACQUIRE, __HIP_MEMORY_SCOPE_AGENT);
    unsigned prev = __hip_atomic_fetch_add(cnt, 1u, __ATOMIC_ACQ_REL, __HIP_MEMORY_SCOPE_AGENT);
    if (prev == nblk - 1u){
      __hip_atomic_store(cnt, 0u, __ATOMIC_RELAXED, __HIP_MEMORY_SCOPE_AGENT);
      __hip_atomic_fetch_add(gen, 1u, __ATOMIC_RELEASE, __HIP_MEMORY_SCOPE_AGENT);
    } else {
      while (__hip_atomic_load(gen, __ATOMIC_ACQUIRE, __HIP_MEMORY_SCOPE_AGENT) == g0)
        __builtin_amdgcn_s_sleep(2);
    }
  }
  __syncthreads();
}

// WMMA 16x32 bf16 tile loader (A: lane=M row, B: lane=N row of W; both row-major, ld=256)
__device__ inline v16bf load16(const bf16_t* base, int row0, int kc, int lane){
  int m = lane & 15, half = lane >> 4;
  const bf16_t* p = base + (size_t)(row0 + m) * E_DIM + kc + half * 8;
  union { v16bf v; uint4 u[2]; } r;
  r.u[0] = *(const uint4*)p;          // K = kc+half*8   .. +7
  r.u[1] = *(const uint4*)(p + 16);   // K = kc+16+half*8.. +7
  return r.v;
}

// ---------- kernels ----------
__global__ void k_init(double* C, unsigned* gEnc, unsigned* colEnc, unsigned* barBuf){
  size_t i = (size_t)blockIdx.x * blockDim.x + threadIdx.x;
  size_t st = (size_t)gridDim.x * blockDim.x;
  for (size_t j = i; j < (size_t)NEXP * BTOK; j += st) C[j] = 1.0;
  for (size_t j = i; j < (size_t)NEXP * N_E;  j += st) colEnc[j] = 0xFFFFFFFFu;
  for (size_t j = i; j < 2u * NEXP;           j += st) gEnc[j] = (j & 1) ? 0u : 0xFFFFFFFFu;
  for (size_t j = i; j < 32u * NEXP;          j += st) barBuf[j] = 0u;
}

// squared norms: wave per row (x rows then emb rows)
__global__ __launch_bounds__(256) void k_norm(const float* x, const float* emb,
                                              float* xnorm, float* wnorm){
  int wave = blockIdx.x * (blockDim.x >> 5) + (threadIdx.x >> 5);
  int lane = threadIdx.x & 31;
  const int NROW = BTOK + NEXP * N_E;
  if (wave >= NROW) return;
  const float* src = (wave < BTOK) ? (x + (size_t)wave * E_DIM)
                                   : (emb + (size_t)(wave - BTOK) * E_DIM);
  const float4* s4 = (const float4*)src;
  float acc = 0.f;
  for (int c = lane; c < E_DIM / 4; c += 32){
    float4 v = s4[c];
    acc += v.x*v.x + v.y*v.y + v.z*v.z + v.w*v.w;
  }
  for (int off = 16; off; off >>= 1) acc += __shfl_xor(acc, off, 32);
  if (lane == 0){
    if (wave < BTOK) xnorm[wave] = acc; else wnorm[wave - BTOK] = acc;
  }
}

// hi/lo bf16 split of x and emb
__global__ void k_cvt(const float* x, const float* emb,
                      bf16_t* xh, bf16_t* xl, bf16_t* wh, bf16_t* wl){
  size_t i = (size_t)blockIdx.x * blockDim.x + threadIdx.x;
  size_t st = (size_t)gridDim.x * blockDim.x;
  const size_t NX = (size_t)BTOK * E_DIM, NW = (size_t)NEXP * N_E * E_DIM;
  for (size_t j = i; j < NX; j += st){
    float v = x[j]; bf16_t h = (bf16_t)v; xh[j] = h; xl[j] = (bf16_t)(v - (float)h);
  }
  for (size_t j = i; j < NW; j += st){
    float v = emb[j]; bf16_t h = (bf16_t)v; wh[j] = h; wl[j] = (bf16_t)(v - (float)h);
  }
}

// distance GEMM: d[b,k] = |x|^2 + |w|^2 - 2 x.w  via split-bf16 WMMA; track min/max + column mins
__global__ __launch_bounds__(256) void k_gemm(const bf16_t* xh, const bf16_t* xl,
    const bf16_t* wh, const bf16_t* wl, const float* xnorm, const float* wnorm,
    float* Dmat, unsigned* gEnc, unsigned* colEnc){
  __shared__ unsigned colLds[32];
  __shared__ unsigned gmnLds, gmxLds;
  int t = threadIdx.x;
  if (t < 32) colLds[t] = 0xFFFFFFFFu;
  if (t == 0){ gmnLds = 0xFFFFFFFFu; gmxLds = 0u; }
  __syncthreads();

  int bid = blockIdx.x;
  int e = bid >> 11;                 // 2048 blocks / expert
  int r = bid & 2047;
  int m0 = (r >> 4) * 64;            // 128 M-blocks of 64
  int n0 = (r & 15) * 32;            // 16 N-blocks of 32
  int w = t >> 5, lane = t & 31;
  int mt = m0 + (w & 3) * 16;
  int nt = n0 + (w >> 2) * 16;
  const bf16_t* whE = wh + (size_t)e * N_E * E_DIM;
  const bf16_t* wlE = wl + (size_t)e * N_E * E_DIM;

  v8f acc = {0.f,0.f,0.f,0.f,0.f,0.f,0.f,0.f};
  for (int kc = 0; kc < E_DIM; kc += 32){
    v16bf ah = load16(xh,  mt, kc, lane);
    v16bf al = load16(xl,  mt, kc, lane);
    v16bf bh = load16(whE, nt, kc, lane);
    v16bf bl = load16(wlE, nt, kc, lane);
    acc = __builtin_amdgcn_wmma_f32_16x16x32_bf16(false, ah, false, bh, (short)0, acc, false, false);
    acc = __builtin_amdgcn_wmma_f32_16x16x32_bf16(false, ah, false, bl, (short)0, acc, false, false);
    acc = __builtin_amdgcn_wmma_f32_16x16x32_bf16(false, al, false, bh, (short)0, acc, false, false);
  }

  int half = lane >> 4;
  int N = nt + (lane & 15);
  float wn = wnorm[e * N_E + N];
  float lmin = 3.4e38f, lmax = -3.4e38f;
  float* dE = Dmat + (size_t)e * BTOK * N_E;
  for (int rr = 0; rr < 8; ++rr){
    int M = mt + rr + half * 8;
    float dv = xnorm[M] + wn - 2.0f * acc[rr];
    dE[(size_t)M * N_E + N] = dv;
    lmin = fminf(lmin, dv); lmax = fmaxf(lmax, dv);
  }
  atomicMin(&colLds[N - n0], f2u(lmin));
  atomicMin(&gmnLds, f2u(lmin));
  atomicMax(&gmxLds, f2u(lmax));
  __syncthreads();
  if (t < 32) atomicMin(&colEnc[e * N_E + n0 + t], colLds[t]);
  if (t == 0){ atomicMin(&gEnc[e*2], gmnLds); atomicMax(&gEnc[e*2+1], gmxLds); }
}

__global__ void k_amp(const unsigned* gEnc, const unsigned* colEnc,
                      float* dminc, float* ampinv){
  int t = threadIdx.x;
  for (int i = t; i < NEXP * N_E; i += blockDim.x) dminc[i] = u2f(colEnc[i]);
  if (t < NEXP){
    float mn = u2f(gEnc[2*t]), mx = u2f(gEnc[2*t+1]);
    float amp = (mx - mn) * 0.5f + 1e-5f;
    ampinv[t] = 1.0f / (amp * 0.01f);   // 1/(amp*eps)
  }
}

// in-place: d -> Q0' = exp(-(d - colmin_k)/(amp*eps))  (row-offset keeps values in (0,1])
__global__ void k_q(float* Dmat, const float* dminc, const float* ampinv){
  size_t i = (size_t)blockIdx.x * blockDim.x + threadIdx.x;
  size_t st = (size_t)gridDim.x * blockDim.x;
  const size_t n4 = (size_t)NEXP * BTOK * N_E / 4;
  float4* q4 = (float4*)Dmat;
  for (size_t j = i; j < n4; j += st){
    size_t el = j * 4;
    int e = (int)(el >> 22);            // 8192*512 = 2^22
    int k = (int)(el & (N_E - 1));
    float4 dm = *(const float4*)(dminc + e * N_E + k);
    float inv = ampinv[e];
    float4 v = q4[j];
    v.x = __expf(-(v.x - dm.x) * inv);
    v.y = __expf(-(v.y - dm.y) * inv);
    v.z = __expf(-(v.z - dm.z) * inv);
    v.w = __expf(-(v.w - dm.w) * inv);
    q4[j] = v;
  }
}

// persistent sinkhorn: scaling-vector form, L2-resident Q0', f64 accumulation
__global__ __launch_bounds__(256) void k_sinkhorn(float* Dmat, double* Cvec, double* Pvec,
    const float* dminc, const float* ampinv, int* indicesAll, unsigned* barBuf){
  __shared__ double lds[512];
  const int e = blockIdx.x >> 4;
  const int g = blockIdx.x & (SKB - 1);
  float*  Qe = Dmat + (size_t)e * BTOK * N_E;
  double* Ce = Cvec + (size_t)e * BTOK;
  double* Pe = Pvec + (size_t)e * N_E;
  unsigned* cnt = barBuf + e * 32;
  unsigned* gen = barBuf + e * 32 + 16;
  const int t  = threadIdx.x;
  const int kk = t & 31, sl = t >> 5, lane = t & 31;
  const int wv = (g * 256 + t) >> 5;          // 0..127 waves per expert

  for (int it = 0; it < SK_ITERS; ++it){
    // row pass: s_k = sum_b Q'[b,k]*C_b ; P_k = 1/(K*s_k)
    {
      const float* qc = Qe + g * 32 + kk;
      double acc = 0.0;
      for (int b = sl; b < BTOK; b += 8)
        acc += (double)qc[(size_t)b * N_E] * Ce[b];
      __syncthreads();
      lds[sl * 32 + kk] = acc;
      __syncthreads();
      if (t < 32){
        double s = 0.0;
        for (int j = 0; j < 8; ++j) s += lds[j * 32 + t];
        s = fmax(s, 1e-300);
        Pe[g * 32 + t] = 1.0 / (512.0 * s);
      }
    }
    gbar(cnt, gen, SKB);
    // col pass: v_b = sum_k P_k*Q'[b,k] ; C_b = 1/(B*v_b)
    {
      lds[t] = Pe[t]; lds[t + 256] = Pe[t + 256];
      __syncthreads();
      for (int bb = 0; bb < 64; ++bb){
        int b = wv * 64 + bb;
        const float4* qr = (const float4*)(Qe + (size_t)b * N_E);
        double acc = 0.0;
        for (int c = 0; c < 4; ++c){
          float4 q = qr[c * 32 + lane];
          int kb = (c * 32 + lane) * 4;
          acc += lds[kb]   * (double)q.x + lds[kb+1] * (double)q.y
               + lds[kb+2] * (double)q.z + lds[kb+3] * (double)q.w;
        }
        for (int off = 16; off; off >>= 1) acc += __shfl_xor(acc, off, 32);
        if (lane == 0) Ce[b] = 1.0 / (8192.0 * fmax(acc, 1e-300));
      }
    }
    gbar(cnt, gen, SKB);
  }
  // argmax_k [log P_k + m_k + log Q'] ; m_k = -colmin_k/(amp*eps)
  {
    float inv = ampinv[e];
    __syncthreads();
    lds[t]       = log(fmax(Pe[t],       1e-300)) - (double)(dminc[e*N_E + t]       * inv);
    lds[t + 256] = log(fmax(Pe[t + 256], 1e-300)) - (double)(dminc[e*N_E + t + 256] * inv);
    __syncthreads();
    for (int bb = 0; bb < 64; ++bb){
      int b = wv * 64 + bb;
      const float4* qr = (const float4*)(Qe + (size_t)b * N_E);
      double best = -1e308; int bi = 1 << 30;
      for (int c = 0; c < 4; ++c){
        float4 q = qr[c * 32 + lane];
        int kb = (c * 32 + lane) * 4;
        float qa[4] = {q.x, q.y, q.z, q.w};
        for (int j = 0; j < 4; ++j){
          int k = kb + j;
          double sc = lds[k] + (qa[j] > 0.f ? log((double)qa[j]) : -1e30);
          if (sc > best || (sc == best && k < bi)){ best = sc; bi = k; }
        }
      }
      for (int off = 16; off; off >>= 1){
        double ob = __shfl_xor(best, off, 32);
        int    oi = __shfl_xor(bi,   off, 32);
        if (ob > best || (ob == best && oi < bi)){ best = ob; bi = oi; }
      }
      if (lane == 0) indicesAll[(size_t)e * BTOK + b] = bi;
    }
  }
}

// gather + gate mixture + straight-through output + mse partials + top-expert index
__global__ __launch_bounds__(256) void k_assemble(const float* x, const float* gate,
    const float* emb, const int* indicesAll, float* xq, float* outXq, float* outIdx,
    double* msePart){
  __shared__ double red[256];
  int b = blockIdx.x, d = threadIdx.x;
  float acc = 0.f;
  for (int e = 0; e < 10; ++e){
    int id = indicesAll[(size_t)e * BTOK + b];
    acc += 0.95f * gate[b * 10 + e] * emb[((size_t)(e * N_E + id)) * E_DIM + d];
  }
  {
    int id = indicesAll[(size_t)10 * BTOK + b];
    acc += 0.05f * emb[((size_t)(10 * N_E + id)) * E_DIM + d];
  }
  float xv = x[(size_t)b * E_DIM + d];
  float diff = acc - xv;
  outXq[(size_t)b * E_DIM + d] = xv + diff;   // straight-through: x + (x_q - x)
  xq[(size_t)b * E_DIM + d] = acc;
  red[d] = (double)diff * (double)diff;
  __syncthreads();
  for (int s = 128; s; s >>= 1){ if (d < s) red[d] += red[d + s]; __syncthreads(); }
  if (d == 0){
    msePart[b] = red[0];
    float bg = -1e38f; int be = 0;
    for (int e = 0; e < 10; ++e){ float g = gate[b * 10 + e]; if (g > bg){ bg = g; be = e; } }
    outIdx[b] = (float)indicesAll[(size_t)be * BTOK + b];
  }
}

// diversity loss per token: -log_softmax(x_q . emb0^T)[y_true], self masked
__global__ __launch_bounds__(256) void k_div(const float* xq, const float* emb,
    const int* label, const int* indicesAll, double* divPart){
  __shared__ float xqs[256];
  __shared__ float sim[512];
  __shared__ float redf[256];
  __shared__ int   redi[256];
  int b = blockIdx.x, t = threadIdx.x;
  xqs[t] = xq[(size_t)b * E_DIM + t];
  __syncthreads();
  int ind0 = indicesAll[b];
  int cluster = label[ind0];
  for (int k = t; k < N_E; k += 256){
    const float* er = emb + (size_t)k * E_DIM;   // codebook 0
    float s = 0.f;
    for (int d2 = 0; d2 < E_DIM; ++d2) s += xqs[d2] * er[d2];
    if (k == ind0) s -= 1e12f;
    sim[k] = s;
  }
  __syncthreads();
  redf[t] = fmaxf(sim[t], sim[t + 256]); __syncthreads();
  for (int s = 128; s; s >>= 1){ if (t < s) redf[t] = fmaxf(redf[t], redf[t + s]); __syncthreads(); }
  float mx = redf[0]; __syncthreads();
  redf[t] = __expf(sim[t] - mx) + __expf(sim[t + 256] - mx);
  int cand = 1 << 30;
  if (label[t] == cluster && t != ind0) cand = t;
  if (label[t + 256] == cluster && (t + 256) != ind0) cand = min(cand, t + 256);
  redi[t] = cand;
  __syncthreads();
  for (int s = 128; s; s >>= 1){
    if (t < s){ redf[t] += redf[t + s]; redi[t] = min(redi[t], redi[t + s]); }
    __syncthreads();
  }
  if (t == 0){
    int y = (redi[0] == (1 << 30)) ? 0 : redi[0];
    float lse = mx + __logf(redf[0]);
    divPart[b] = (double)(lse - sim[y]);
  }
}

__global__ void k_final(const double* msePart, const double* divPart, float* outLoss){
  __shared__ double red[256];
  int t = threadIdx.x;
  double a = 0.0, dv = 0.0;
  for (int i = t; i < BTOK; i += 256){ a += msePart[i]; dv += divPart[i]; }
  red[t] = a; __syncthreads();
  for (int s = 128; s; s >>= 1){ if (t < s) red[t] += red[t + s]; __syncthreads(); }
  double mse = red[0]; __syncthreads();
  red[t] = dv; __syncthreads();
  for (int s = 128; s; s >>= 1){ if (t < s) red[t] += red[t + s]; __syncthreads(); }
  if (t == 0){
    mse /= (double)BTOK * E_DIM;
    double div = red[0] / (double)BTOK;
    outLoss[0] = (float)(1.25 * mse + div);   // codebook + MU*commitment + BETA*div
  }
}

// ---------- launch ----------
extern "C" void kernel_launch(void* const* d_in, const int* in_sizes, int n_in,
                              void* d_out, int out_size, void* d_ws, size_t ws_size,
                              hipStream_t stream) {
  const float* x     = (const float*)d_in[0];
  const int*   label = (const int*)d_in[1];
  const float* gate  = (const float*)d_in[3];
  const float* emb   = (const float*)d_in[4];
  (void)in_sizes; (void)n_in; (void)out_size; (void)ws_size;

  char* ws = (char*)d_ws;
  size_t off = 0;
  auto alloc = [&](size_t bytes) -> char* {
    char* p = ws + off;
    off = (off + bytes + 255) & ~(size_t)255;
    return p;
  };
  float*    Dmat    = (float*)   alloc((size_t)NEXP * BTOK * N_E * 4);
  bf16_t*   xh      = (bf16_t*)  alloc((size_t)BTOK * E_DIM * 2);
  bf16_t*   xl      = (bf16_t*)  alloc((size_t)BTOK * E_DIM * 2);
  bf16_t*   wh      = (bf16_t*)  alloc((size_t)NEXP * N_E * E_DIM * 2);
  bf16_t*   wl      = (bf16_t*)  alloc((size_t)NEXP * N_E * E_DIM * 2);
  float*    xnorm   = (float*)   alloc((size_t)BTOK * 4);
  float*    wnorm   = (float*)   alloc((size_t)NEXP * N_E * 4);
  unsigned* gEnc    = (unsigned*)alloc(2u * NEXP * 4);
  unsigned* colEnc  = (unsigned*)alloc((size_t)NEXP * N_E * 4);
  float*    dminc   = (float*)   alloc((size_t)NEXP * N_E * 4);
  float*    ampinv  = (float*)   alloc(NEXP * 4);
  double*   Cvec    = (double*)  alloc((size_t)NEXP * BTOK * 8);
  double*   Pvec    = (double*)  alloc((size_t)NEXP * N_E * 8);
  int*      indAll  = (int*)     alloc((size_t)NEXP * BTOK * 4);
  float*    xq      = (float*)   alloc((size_t)BTOK * E_DIM * 4);
  double*   msePart = (double*)  alloc((size_t)BTOK * 8);
  double*   divPart = (double*)  alloc((size_t)BTOK * 8);
  unsigned* barBuf  = (unsigned*)alloc(32u * NEXP * 4);

  float* outXq   = (float*)d_out;
  float* outLoss = outXq + (size_t)BTOK * E_DIM;
  float* outIdx  = outLoss + 1;

  k_init    <<<256, 256, 0, stream>>>(Cvec, gEnc, colEnc, barBuf);
  k_norm    <<<(BTOK + NEXP * N_E) / 8, 256, 0, stream>>>(x, emb, xnorm, wnorm);
  k_cvt     <<<2048, 256, 0, stream>>>(x, emb, xh, xl, wh, wl);
  k_gemm    <<<NEXP * 2048, 256, 0, stream>>>(xh, xl, wh, wl, xnorm, wnorm, Dmat, gEnc, colEnc);
  k_amp     <<<1, 256, 0, stream>>>(gEnc, colEnc, dminc, ampinv);
  k_q       <<<4096, 256, 0, stream>>>(Dmat, dminc, ampinv);
  k_sinkhorn<<<NEXP * SKB, 256, 0, stream>>>(Dmat, Cvec, Pvec, dminc, ampinv, indAll, barBuf);
  k_assemble<<<BTOK, 256, 0, stream>>>(x, gate, emb, indAll, xq, outXq, outIdx, msePart);
  k_div     <<<BTOK, 256, 0, stream>>>(xq, emb, label, indAll, divPart);
  k_final   <<<1, 256, 0, stream>>>(msePart, divPart, outLoss);
}
